// DINOv3_Adapter_86474871538107
// MI455X (gfx1250) — compile-verified, matrix operand order checked by reading, not run on hardware
//
#include <hip/hip_runtime.h>
#include <hip/hip_bf16.h>

typedef __attribute__((ext_vector_type(16))) __bf16 v16bf;
typedef __attribute__((ext_vector_type(8)))  __bf16 v8bf;
typedef __attribute__((ext_vector_type(8)))  float  v8f;

#define B_      4
#define D_      384
#define HEADS_  6
#define POINTS_ 4
#define TH_     32
#define TW_     32
#define L1_     4096
#define L2_     1024
#define L3_     256
#define LQ_     5376
#define HID_    96
#define DH_     64
#define EPS_    1e-6f

#define NQ_ (B_ * LQ_)          // 21504 query tokens
#define NF_ (B_ * TH_ * TW_)    // 4096 feat tokens

// ---------------------------------------------------------------------------
// Load one 16x32 bf16 fragment in WMMA A/B lane layout:
// lane 0-15 = row (lane&15); chunks K {koff..koff+7} and {16+koff..} with
// koff = 8*(lane>=16). p already points at row start + koff.
// ---------------------------------------------------------------------------
__device__ __forceinline__ v16bf load_frag(const __bf16* __restrict__ p) {
  v8bf lo = *(const v8bf*)p;
  v8bf hi = *(const v8bf*)(p + 16);
  v16bf r;
#pragma unroll
  for (int i = 0; i < 8; ++i) { r[i] = lo[i]; r[i + 8] = hi[i]; }
  return r;
}

// ---------------------------------------------------------------------------
// LayerNorm over D=384, one wave (32 lanes) per token, writes bf16 for WMMA A.
// ---------------------------------------------------------------------------
__global__ void ln_bf16_kernel(const float* __restrict__ x,
                               const float* __restrict__ g,
                               const float* __restrict__ b,
                               __bf16* __restrict__ out, int ntok) {
  int wid  = (blockIdx.x * blockDim.x + threadIdx.x) >> 5;
  int lane = threadIdx.x & 31;
  if (wid >= ntok) return;
  const float* xr = x + (size_t)wid * D_;
  float v[12];
  float s = 0.f, ss = 0.f;
  for (int i = 0; i < 12; ++i) {
    float t = xr[lane + 32 * i];
    v[i] = t; s += t; ss += t * t;
  }
  for (int m = 16; m >= 1; m >>= 1) {
    s  += __shfl_xor(s,  m, 32);
    ss += __shfl_xor(ss, m, 32);
  }
  float mean = s * (1.f / D_);
  float var  = ss * (1.f / D_) - mean * mean;
  float rs   = rsqrtf(var + EPS_);
  __bf16* outr = out + (size_t)wid * D_;
  for (int i = 0; i < 12; ++i) {
    int c = lane + 32 * i;
    outr[c] = (__bf16)((v[i] - mean) * rs * g[c] + b[c]);
  }
}

// ---------------------------------------------------------------------------
// Weight prep: W (K x N, f32) -> Bt (Npad x K, bf16), zero-padded rows.
// ---------------------------------------------------------------------------
__global__ void wtrans_kernel(const float* __restrict__ W, __bf16* __restrict__ Bt,
                              int K, int N, int Npad) {
  int idx = blockIdx.x * blockDim.x + threadIdx.x;
  if (idx >= Npad * K) return;
  int n = idx / K, k = idx - n * K;
  float v = (n < N) ? W[(size_t)k * N + n] : 0.f;
  Bt[idx] = (__bf16)v;
}

// ---------------------------------------------------------------------------
// Generic bf16 WMMA GEMM (16x16 tile per wave) — used for narrow-N GEMMs
// (off: N=48, att: N=24 with padded Bt). C = A * Bt^T + bias (+res).
// ---------------------------------------------------------------------------
__global__ void gemm_bf16_kernel(const __bf16* __restrict__ A,
                                 const __bf16* __restrict__ Bt,
                                 const float* __restrict__ bias,
                                 const float* __restrict__ res,
                                 float* __restrict__ C,
                                 int M, int N, int K, int tilesN) {
  int lane = threadIdx.x & 31;
  int wave = (blockIdx.x * blockDim.x + threadIdx.x) >> 5;
  int tilesM = M >> 4;
  if (wave >= tilesM * tilesN) return;  // wave-uniform exit (EXEC all-1s)
  int tm = wave / tilesN;
  int tn = wave - tm * tilesN;
  int m0 = tm << 4, n0 = tn << 4;
  int r    = lane & 15;
  int koff = (lane >> 4) << 3;
  const __bf16* arow = A  + (size_t)(m0 + r) * K + koff;
  const __bf16* brow = Bt + (size_t)(n0 + r) * K + koff;
  v8f acc = {};
  for (int kk = 0; kk < K; kk += 32) {
    v16bf av = load_frag(arow + kk);
    v16bf bv = load_frag(brow + kk);
    acc = __builtin_amdgcn_wmma_f32_16x16x32_bf16(
        false, av, false, bv, (short)0, acc, false, false);
  }
  int n = n0 + r;
  if (n >= N) return;  // padded N tiles only
  int mb = m0 + ((lane >> 4) << 3);
  float bn = bias ? bias[n] : 0.f;
#pragma unroll
  for (int j = 0; j < 8; ++j) {
    size_t o = (size_t)(mb + j) * N + n;
    float val = acc[j] + bn;
    if (res) val += res[o];
    C[o] = val;
  }
}

// ---------------------------------------------------------------------------
// Register-blocked bf16 WMMA GEMM: 32x32 tile per wave, 4 accumulators.
// Per K-step: 4 fragment loads -> 4 WMMAs (2x the arithmetic intensity of the
// 16x16 kernel). Requires M%32==0 and N%32==0 (true for all big GEMMs here).
// ---------------------------------------------------------------------------
__global__ void gemm_bf16_32x32_kernel(const __bf16* __restrict__ A,
                                       const __bf16* __restrict__ Bt,
                                       const float* __restrict__ bias,
                                       const float* __restrict__ res,
                                       float* __restrict__ C,
                                       int M, int N, int K, int tilesN) {
  int lane = threadIdx.x & 31;
  int wave = (blockIdx.x * blockDim.x + threadIdx.x) >> 5;
  int tilesM = M >> 5;
  if (wave >= tilesM * tilesN) return;  // wave-uniform exit
  int tm = wave / tilesN;
  int tn = wave - tm * tilesN;
  int m0 = tm << 5, n0 = tn << 5;
  int r    = lane & 15;
  int koff = (lane >> 4) << 3;
  const __bf16* arow0 = A  + (size_t)(m0 + r) * K + koff;
  const __bf16* arow1 = arow0 + (size_t)16 * K;
  const __bf16* brow0 = Bt + (size_t)(n0 + r) * K + koff;
  const __bf16* brow1 = brow0 + (size_t)16 * K;
  v8f acc00 = {}, acc01 = {}, acc10 = {}, acc11 = {};
  for (int kk = 0; kk < K; kk += 32) {
    v16bf a0 = load_frag(arow0 + kk);
    v16bf a1 = load_frag(arow1 + kk);
    v16bf b0 = load_frag(brow0 + kk);
    v16bf b1 = load_frag(brow1 + kk);
    acc00 = __builtin_amdgcn_wmma_f32_16x16x32_bf16(false, a0, false, b0, (short)0, acc00, false, false);
    acc01 = __builtin_amdgcn_wmma_f32_16x16x32_bf16(false, a0, false, b1, (short)0, acc01, false, false);
    acc10 = __builtin_amdgcn_wmma_f32_16x16x32_bf16(false, a1, false, b0, (short)0, acc10, false, false);
    acc11 = __builtin_amdgcn_wmma_f32_16x16x32_bf16(false, a1, false, b1, (short)0, acc11, false, false);
  }
  // store 4 subtiles; C/D layout: VGPR j -> M = sub_m0 + 8*(lane>=16)+j,
  //                               N = sub_n0 + (lane&15)
  int mhalf = (lane >> 4) << 3;
  float bn0 = bias ? bias[n0 + r]      : 0.f;
  float bn1 = bias ? bias[n0 + 16 + r] : 0.f;
  const v8f* accs[4] = { &acc00, &acc01, &acc10, &acc11 };
#pragma unroll
  for (int mi = 0; mi < 2; ++mi) {
#pragma unroll
    for (int ni = 0; ni < 2; ++ni) {
      const v8f& a = *accs[mi * 2 + ni];
      int n  = n0 + ni * 16 + r;
      int mb = m0 + mi * 16 + mhalf;
      float bn = ni ? bn1 : bn0;
#pragma unroll
      for (int j = 0; j < 8; ++j) {
        size_t o = (size_t)(mb + j) * N + n;
        float val = a[j] + bn;
        if (res) val += res[o];
        C[o] = val;
      }
    }
  }
}

// ---------------------------------------------------------------------------
// Deformable sampling: per (b, q, head): softmax over 4 points, 4 bilinear
// samples of the 64-ch head slice of value, weighted sum. 64 threads/group.
// ---------------------------------------------------------------------------
__global__ void msdeform_kernel(const float* __restrict__ value,  // (B,1024,384)
                                const float* __restrict__ off,    // (B*LQ,48)
                                const float* __restrict__ attL,   // (B*LQ,24)
                                const float* __restrict__ ref,    // (LQ,2)
                                __bf16* __restrict__ out) {       // (B*LQ,384)
  int gid = blockIdx.x * 4 + (threadIdx.x >> 6);
  int c   = threadIdx.x & 63;
  if (gid >= B_ * LQ_ * HEADS_) return;
  int h = gid % HEADS_;
  int t = gid / HEADS_;   // b*LQ + q
  int q = t % LQ_;
  int b = t / LQ_;

  const float* offp = off  + (size_t)t * (HEADS_ * POINTS_ * 2) + h * (POINTS_ * 2);
  const float* attp = attL + (size_t)t * (HEADS_ * POINTS_) + h * POINTS_;
  float rx = ref[q * 2 + 0], ry = ref[q * 2 + 1];

  float lg[4];
  float mx = -1e30f;
#pragma unroll
  for (int p = 0; p < 4; ++p) { lg[p] = attp[p]; mx = fmaxf(mx, lg[p]); }
  float den = 0.f;
#pragma unroll
  for (int p = 0; p < 4; ++p) { lg[p] = __expf(lg[p] - mx); den += lg[p]; }
  float inv = 1.f / den;

  const float* vb = value + (size_t)b * (TH_ * TW_) * D_ + h * DH_ + c;
  float acc = 0.f;
#pragma unroll
  for (int p = 0; p < 4; ++p) {
    float x = (rx + offp[p * 2 + 0] * (1.f / TW_)) * TW_ - 0.5f;
    float y = (ry + offp[p * 2 + 1] * (1.f / TH_)) * TH_ - 0.5f;
    float x0f = floorf(x), y0f = floorf(y);
    float fx = x - x0f, fy = y - y0f;
    int x0 = (int)x0f, y0 = (int)y0f;
    float s = 0.f;
#pragma unroll
    for (int dy = 0; dy < 2; ++dy) {
#pragma unroll
      for (int dx = 0; dx < 2; ++dx) {
        int ix = x0 + dx, iy = y0 + dy;
        bool valid = (ix >= 0) && (ix < TW_) && (iy >= 0) && (iy < TH_);
        int cx = ix < 0 ? 0 : (ix > TW_ - 1 ? TW_ - 1 : ix);
        int cy = iy < 0 ? 0 : (iy > TH_ - 1 ? TH_ - 1 : iy);
        float g = vb[(size_t)(cy * TW_ + cx) * D_];
        float w = (dx ? fx : 1.f - fx) * (dy ? fy : 1.f - fy);
        s += valid ? g * w : 0.f;
      }
    }
    acc += lg[p] * inv * s;
  }
  out[(size_t)t * D_ + h * DH_ + c] = (__bf16)acc;
}

// ---------------------------------------------------------------------------
// Depthwise 3x3 conv (SAME, 3 pyramid levels) + bias + exact GELU -> bf16.
// ---------------------------------------------------------------------------
__global__ void dwconv_gelu_kernel(const float* __restrict__ y1,   // (B*LQ,HID)
                                   const float* __restrict__ k,    // (HID,1,3,3)
                                   const float* __restrict__ kb,   // (HID)
                                   __bf16* __restrict__ out) {     // (B*LQ,HID)
  int idx = blockIdx.x * blockDim.x + threadIdx.x;
  if (idx >= B_ * LQ_ * HID_) return;
  int c   = idx % HID_;
  int t   = idx / HID_;
  int pos = t % LQ_;
  int b   = t / LQ_;
  int h, w, base, rel;
  if (pos < L1_)             { h = 2 * TH_; w = 2 * TW_; base = 0;          rel = pos; }
  else if (pos < L1_ + L2_)  { h = TH_;     w = TW_;     base = L1_;        rel = pos - L1_; }
  else                       { h = TH_ / 2; w = TW_ / 2; base = L1_ + L2_;  rel = pos - L1_ - L2_; }
  int yy = rel / w, xx = rel - yy * w;
  const float* kc = k + c * 9;
  float acc = kb[c];
#pragma unroll
  for (int dy = -1; dy <= 1; ++dy) {
    int ny = yy + dy;
    if (ny < 0 || ny >= h) continue;
#pragma unroll
    for (int dx = -1; dx <= 1; ++dx) {
      int nx = xx + dx;
      if (nx < 0 || nx >= w) continue;
      acc += y1[((size_t)b * LQ_ + base + ny * w + nx) * HID_ + c] * kc[(dy + 1) * 3 + (dx + 1)];
    }
  }
  float gl = 0.5f * acc * (1.f + erff(acc * 0.70710678118654752f));
  out[idx] = (__bf16)gl;
}

// ---------------------------------------------------------------------------
extern "C" void kernel_launch(void* const* d_in, const int* in_sizes, int n_in,
                              void* d_out, int out_size, void* d_ws, size_t ws_size,
                              hipStream_t stream) {
  const float* query = (const float*)d_in[0];
  const float* feat  = (const float*)d_in[1];
  const float* refp  = (const float*)d_in[2];
  const float* qn_g  = (const float*)d_in[3];
  const float* qn_b  = (const float*)d_in[4];
  const float* fn_g  = (const float*)d_in[5];
  const float* fn_b  = (const float*)d_in[6];
  const float* W_val = (const float*)d_in[7];
  const float* b_val = (const float*)d_in[8];
  const float* W_off = (const float*)d_in[9];
  const float* b_off = (const float*)d_in[10];
  const float* W_att = (const float*)d_in[11];
  const float* b_att = (const float*)d_in[12];
  const float* W_out = (const float*)d_in[13];
  const float* b_out = (const float*)d_in[14];
  const float* ffn_g = (const float*)d_in[15];
  const float* ffn_b = (const float*)d_in[16];
  const float* W_fc1 = (const float*)d_in[17];
  const float* b_fc1 = (const float*)d_in[18];
  const float* dw_k  = (const float*)d_in[19];
  const float* dw_b  = (const float*)d_in[20];
  const float* W_fc2 = (const float*)d_in[21];
  const float* b_fc2 = (const float*)d_in[22];
  float* outp = (float*)d_out;

  // --- workspace carve (256B aligned) ---
  char* p = (char*)d_ws;
  auto carve = [&](size_t bytes) -> void* {
    void* r = (void*)p;
    p += (bytes + 255) & ~(size_t)255;
    return r;
  };
  __bf16* qln     = (__bf16*)carve((size_t)NQ_ * D_ * 2);
  __bf16* fln     = (__bf16*)carve((size_t)NF_ * D_ * 2);
  float*  value   = (float*) carve((size_t)NF_ * D_ * 4);
  float*  offbuf  = (float*) carve((size_t)NQ_ * 48 * 4);
  float*  attbuf  = (float*) carve((size_t)NQ_ * 24 * 4);
  __bf16* sampled = (__bf16*)carve((size_t)NQ_ * D_ * 2);
  float*  q2      = (float*) carve((size_t)NQ_ * D_ * 4);
  __bf16* ffnln   = (__bf16*)carve((size_t)NQ_ * D_ * 2);
  float*  y1      = (float*) carve((size_t)NQ_ * HID_ * 4);
  __bf16* y2      = (__bf16*)carve((size_t)NQ_ * HID_ * 2);
  __bf16* Wval_t  = (__bf16*)carve((size_t)384 * 384 * 2);
  __bf16* Woff_t  = (__bf16*)carve((size_t)48  * 384 * 2);
  __bf16* Watt_t  = (__bf16*)carve((size_t)32  * 384 * 2);  // 24 padded to 32
  __bf16* Wout_t  = (__bf16*)carve((size_t)384 * 384 * 2);
  __bf16* Wfc1_t  = (__bf16*)carve((size_t)96  * 384 * 2);
  __bf16* Wfc2_t  = (__bf16*)carve((size_t)384 * 96  * 2);

  auto wtrans = [&](const float* W, __bf16* Bt, int K, int N, int Npad) {
    int n = Npad * K;
    wtrans_kernel<<<(n + 255) / 256, 256, 0, stream>>>(W, Bt, K, N, Npad);
  };
  // Register-blocked GEMM for M%32==0 && N%32==0, generic 16x16 otherwise.
  auto gemm = [&](const __bf16* A, const __bf16* Bt, const float* bias,
                  const float* res, float* C, int M, int N, int K) {
    if ((M & 31) == 0 && (N & 31) == 0) {
      int tilesN = N >> 5;
      int waves  = (M >> 5) * tilesN;
      gemm_bf16_32x32_kernel<<<(waves + 7) / 8, 256, 0, stream>>>(A, Bt, bias, res, C, M, N, K, tilesN);
    } else {
      int tilesN = (N + 15) >> 4;
      int waves  = (M >> 4) * tilesN;
      gemm_bf16_kernel<<<(waves + 7) / 8, 256, 0, stream>>>(A, Bt, bias, res, C, M, N, K, tilesN);
    }
  };

  // weight prep (bf16, transposed)
  wtrans(W_val, Wval_t, D_, D_,   D_);
  wtrans(W_off, Woff_t, D_, 48,   48);
  wtrans(W_att, Watt_t, D_, 24,   32);
  wtrans(W_out, Wout_t, D_, D_,   D_);
  wtrans(W_fc1, Wfc1_t, D_, HID_, HID_);
  wtrans(W_fc2, Wfc2_t, HID_, D_, D_);

  // LayerNorms -> bf16 activations
  ln_bf16_kernel<<<(NQ_ + 7) / 8, 256, 0, stream>>>(query, qn_g, qn_b, qln, NQ_);
  ln_bf16_kernel<<<(NF_ + 7) / 8, 256, 0, stream>>>(feat,  fn_g, fn_b, fln, NF_);

  // value = LN(feat) @ W_val + b_val            (32x32 blocked)
  gemm(fln, Wval_t, b_val, nullptr, value, NF_, D_, D_);
  // off / att logits from LN(query)             (16x16 generic, narrow N)
  gemm(qln, Woff_t, b_off, nullptr, offbuf, NQ_, 48, D_);
  gemm(qln, Watt_t, b_att, nullptr, attbuf, NQ_, 24, D_);

  // deformable sampling -> bf16 (B*LQ, D)
  {
    int groups = B_ * LQ_ * HEADS_;
    msdeform_kernel<<<(groups + 3) / 4, 256, 0, stream>>>(value, offbuf, attbuf, refp, sampled);
  }

  // q2 = sampled @ W_out + b_out + query        (32x32 blocked, residual fused)
  gemm(sampled, Wout_t, b_out, query, q2, NQ_, D_, D_);

  // FFN: LN(q2) -> fc1 -> dwconv+gelu -> fc2 (+q2 residual) -> d_out
  ln_bf16_kernel<<<(NQ_ + 7) / 8, 256, 0, stream>>>(q2, ffn_g, ffn_b, ffnln, NQ_);
  gemm(ffnln, Wfc1_t, b_fc1, nullptr, y1, NQ_, HID_, D_);
  {
    int n = B_ * LQ_ * HID_;
    dwconv_gelu_kernel<<<(n + 255) / 256, 256, 0, stream>>>(y1, dw_k, dw_b, y2);
  }
  gemm(y2, Wfc2_t, b_fc2, q2, outp, NQ_, D_, HID_);
}
